// WormholeAttentionBlock_29222957481986
// MI455X (gfx1250) — matrix operand order repaired and checked by reading, not compiled, not run on hardware
//
#include <hip/hip_runtime.h>
#include <hip/hip_bf16.h>
#include <math.h>

// ---------------------------------------------------------------------------
// WormholeAttentionBlock for MI455X (gfx1250, wave32, WMMA bf16 + async LDS)
// ---------------------------------------------------------------------------
typedef __attribute__((ext_vector_type(16))) __bf16 v16bf;
typedef __attribute__((ext_vector_type(8)))  float  v8f;
typedef __attribute__((ext_vector_type(8)))  __bf16 bf16x8;
typedef __attribute__((ext_vector_type(4)))  int    v4i;

#define B_   8
#define S_   577
#define D_   768
#define H_   12
#define HD_  64
#define P_   576
#define KW_  32
#define NS_  (B_*S_)     // 4616 tokens
#define D3_  (3*D_)      // 2304
#define DM_  (4*D_)      // 3072
#define SCALE_ 0.125f    // HD^-0.5
#define INV_TEMP_ 10.0f

#define AS1 __attribute__((address_space(1)))
#define AS3 __attribute__((address_space(3)))

#if __has_builtin(__builtin_amdgcn_global_load_async_to_lds_b128) && \
    __has_builtin(__builtin_amdgcn_s_wait_asynccnt)
#define WH_ASYNC 1
#else
#define WH_ASYNC 0
#endif

// ---------------------------------------------------------------------------
// Transpose + convert: in (K x N) f32 row-major -> out (N x K) bf16 row-major
// ---------------------------------------------------------------------------
__global__ __launch_bounds__(256) void transpose_bf16_kernel(
    const float* __restrict__ in, __bf16* __restrict__ out, int K, int N) {
    __shared__ float tile[32][33];
    const int k0 = blockIdx.y * 32, n0 = blockIdx.x * 32;
    const int tx = threadIdx.x, ty = threadIdx.y;   // 32 x 8
#pragma unroll
    for (int i = 0; i < 32; i += 8)
        tile[ty + i][tx] = in[(long long)(k0 + ty + i) * N + n0 + tx];
    __syncthreads();
#pragma unroll
    for (int i = 0; i < 32; i += 8)
        out[(long long)(n0 + ty + i) * K + k0 + tx] = (__bf16)tile[tx][ty + i];
}

// ---------------------------------------------------------------------------
// f32 -> bf16 conversion
// ---------------------------------------------------------------------------
__global__ void f32_to_bf16_kernel(const float* __restrict__ in,
                                   __bf16* __restrict__ out, long long n) {
    long long i = (long long)blockIdx.x * 256 + threadIdx.x;
    if (i < n) out[i] = (__bf16)in[i];
}

// ---------------------------------------------------------------------------
// LayerNorm over D=768 (one block of 256 threads per token row)
// ---------------------------------------------------------------------------
__global__ __launch_bounds__(256) void layernorm_kernel(
    const float* __restrict__ x, const float* __restrict__ g,
    const float* __restrict__ b, float* __restrict__ of,
    __bf16* __restrict__ obf) {
    const long long row = blockIdx.x;
    const int tid = threadIdx.x;
    const float* xr = x + row * D_;
    float vals[3];
    float s = 0.f, s2 = 0.f;
#pragma unroll
    for (int i = 0; i < 3; ++i) {
        float v = xr[tid + i * 256];
        vals[i] = v; s += v; s2 += v * v;
    }
#pragma unroll
    for (int o = 16; o > 0; o >>= 1) { s += __shfl_xor(s, o); s2 += __shfl_xor(s2, o); }
    __shared__ float red0[8], red1[8];
    const int wave = tid >> 5, lane = tid & 31;
    if (lane == 0) { red0[wave] = s; red1[wave] = s2; }
    __syncthreads();
    if (wave == 0) {
        s = red0[lane & 7]; s2 = red1[lane & 7];
#pragma unroll
        for (int o = 4; o > 0; o >>= 1) { s += __shfl_xor(s, o); s2 += __shfl_xor(s2, o); }
        if (lane == 0) { red0[0] = s; red1[0] = s2; }
    }
    __syncthreads();
    const float mean = red0[0] * (1.0f / D_);
    const float var  = red1[0] * (1.0f / D_) - mean * mean;
    const float rstd = rsqrtf(var + 1e-5f);
#pragma unroll
    for (int i = 0; i < 3; ++i) {
        const int c = tid + i * 256;
        const float v = (vals[i] - mean) * rstd * g[c] + b[c];
        if (of)  of[row * D_ + c] = v;
        if (obf) obf[row * D_ + c] = (__bf16)v;
    }
}

// ---------------------------------------------------------------------------
// L2-normalize rows of full (B,S,D), dropping s==0, pack to (B,P,D) bf16
// ---------------------------------------------------------------------------
__global__ __launch_bounds__(256) void l2norm_pack_kernel(
    const float* __restrict__ full, __bf16* __restrict__ outbf) {
    const long long rp = blockIdx.x;               // 0..B*P-1
    const long long b = rp / P_, pp = rp % P_;
    const float* xr = full + (b * S_ + pp + 1) * D_;
    const int tid = threadIdx.x;
    float vals[3];
    float s2 = 0.f;
#pragma unroll
    for (int i = 0; i < 3; ++i) { float v = xr[tid + i * 256]; vals[i] = v; s2 += v * v; }
#pragma unroll
    for (int o = 16; o > 0; o >>= 1) s2 += __shfl_xor(s2, o);
    __shared__ float red[8];
    const int wave = tid >> 5, lane = tid & 31;
    if (lane == 0) red[wave] = s2;
    __syncthreads();
    float tot = red[0] + red[1] + red[2] + red[3] + red[4] + red[5] + red[6] + red[7];
    const float inv = 1.0f / fmaxf(sqrtf(tot), 1e-12f);
#pragma unroll
    for (int i = 0; i < 3; ++i) {
        const int c = tid + i * 256;
        outbf[rp * D_ + c] = (__bf16)(vals[i] * inv);
    }
}

// ---------------------------------------------------------------------------
// bf16 WMMA GEMM (NT form):  C[M,N] = A[M,K] @ Bt[N,K]^T  (+bias, +epilogue)
//   EPI: 0 = bias, 1 = bias + exact GELU, 2 = bias + residual
// Block = 256 threads = 8 waves as 2(M) x 4(N).
// Block tile (MT*32) x (NT*64); per-wave (MT*16) x (NT*16), MT*NT wmma/K-step.
// K-step 32.  A/B tiles staged via GLOBAL_LOAD_ASYNC_TO_LDS_B128 (ASYNCcnt),
// falling back to sync loads when the builtin is absent.
// ---------------------------------------------------------------------------
template <int MT, int NT, int EPI>
__global__ __launch_bounds__(256) void gemm_bt_wmma(
    const __bf16* __restrict__ A, const __bf16* __restrict__ Bt,
    const float* __restrict__ bias, const float* __restrict__ residual,
    float* __restrict__ C, __bf16* __restrict__ Cbf,
    int M, int N, int K, long long sA, long long sB, long long sC) {
    constexpr int BM = MT * 32;
    constexpr int BN = NT * 64;
    __shared__ __bf16 As[BM][40];
    __shared__ __bf16 Bs[BN][40];

    const int tid = threadIdx.x;
    const long long z = blockIdx.z;
    A  += z * sA;
    Bt += z * sB;
    const long long cbase = z * sC;
    const int m0 = blockIdx.y * BM;
    const int n0 = blockIdx.x * BN;
    const int wave = tid >> 5, lane = tid & 31;
    const int wm = wave >> 2, wn = wave & 3;        // 2 x 4 wave grid
    const int row16 = lane & 15, half = lane >> 4;

    // A staging: one b128 (8 bf16) per thread; MT==1 needs only 128 threads
    const int ar = tid >> 2, ak = (tid & 3) * 8;
    const bool aact = (MT == 2) ? true : (tid < 128);
    // B staging coords
    const int br2 = tid >> 1, bk2 = (tid & 1) * 16;   // NT==2: 2 x b128
    const int br1 = tid >> 2, bk1 = (tid & 3) * 8;    // NT==1: 1 x b128

    __bf16* Anc  = const_cast<__bf16*>(A);
    __bf16* Btnc = const_cast<__bf16*>(Bt);

    v8f acc[MT][NT] = {};

    for (int k0 = 0; k0 < K; k0 += 32) {
#if WH_ASYNC
        if (aact && (m0 + ar < M))
            __builtin_amdgcn_global_load_async_to_lds_b128(
                (AS1 v4i*)(Anc + (long long)(m0 + ar) * K + k0 + ak),
                (AS3 v4i*)&As[ar][ak], 0, 0);
        if (NT == 2) {
            __builtin_amdgcn_global_load_async_to_lds_b128(
                (AS1 v4i*)(Btnc + (long long)(n0 + br2) * K + k0 + bk2),
                (AS3 v4i*)&Bs[br2][bk2], 0, 0);
            __builtin_amdgcn_global_load_async_to_lds_b128(
                (AS1 v4i*)(Btnc + (long long)(n0 + br2) * K + k0 + bk2 + 8),
                (AS3 v4i*)&Bs[br2][bk2 + 8], 0, 0);
        } else {
            __builtin_amdgcn_global_load_async_to_lds_b128(
                (AS1 v4i*)(Btnc + (long long)(n0 + br1) * K + k0 + bk1),
                (AS3 v4i*)&Bs[br1][bk1], 0, 0);
        }
        __builtin_amdgcn_s_wait_asynccnt(0);
        __syncthreads();
#else
        {
            bf16x8 av = {};
            if (aact && (m0 + ar < M))
                av = *(const bf16x8*)(A + (long long)(m0 + ar) * K + k0 + ak);
            if (aact) *(bf16x8*)(&As[ar][ak]) = av;
            if (NT == 2) {
                const __bf16* bp = Bt + (long long)(n0 + br2) * K + k0 + bk2;
                *(bf16x8*)(&Bs[br2][bk2])     = *(const bf16x8*)(bp);
                *(bf16x8*)(&Bs[br2][bk2 + 8]) = *(const bf16x8*)(bp + 8);
            } else {
                *(bf16x8*)(&Bs[br1][bk1]) =
                    *(const bf16x8*)(Bt + (long long)(n0 + br1) * K + k0 + bk1);
            }
            if (k0 + 32 < K) {
                if (aact && (m0 + ar < M))
                    __builtin_prefetch(A + (long long)(m0 + ar) * K + k0 + 32 + ak, 0, 3);
                __builtin_prefetch(Bt + (long long)(n0 + br1) * K + k0 + 32 + bk1, 0, 3);
            }
        }
        __syncthreads();
#endif

        // ---- fragments per ISA VGPR layouts (CDNA5 ISA 7.12.2) ----
        v16bf af[MT], bfv[NT];
#pragma unroll
        for (int t = 0; t < MT; ++t)
#pragma unroll
            for (int e = 0; e < 16; ++e) {
                const int v = e >> 1, hi = e & 1;
                const int kA = ((v & 4) << 2) + half * 8 + ((v & 3) << 1) + hi;
                af[t][e] = As[wm * (MT * 16) + t * 16 + row16][kA];
            }
#pragma unroll
        for (int u = 0; u < NT; ++u)
#pragma unroll
            for (int e = 0; e < 16; ++e)
                bfv[u][e] = Bs[wn * (NT * 16) + u * 16 + row16][((e & 8) << 1) + half * 8 + (e & 7)];

#pragma unroll
        for (int t = 0; t < MT; ++t)
#pragma unroll
            for (int u = 0; u < NT; ++u)
                acc[t][u] = __builtin_amdgcn_wmma_f32_16x16x32_bf16(
                    false, af[t], false, bfv[u], (short)0, acc[t][u], false, false);
        __syncthreads();
    }

    // ---- epilogue: C/D layout -> vgpr r holds M = half*8 + r, N = lane&15 ----
#pragma unroll
    for (int t = 0; t < MT; ++t) {
#pragma unroll
        for (int u = 0; u < NT; ++u) {
            const int cn = n0 + wn * (NT * 16) + u * 16 + row16;
            const float bv = bias ? bias[cn] : 0.0f;
#pragma unroll
            for (int r = 0; r < 8; ++r) {
                const int cm = m0 + wm * (MT * 16) + t * 16 + half * 8 + r;
                if (cm < M) {
                    float v = acc[t][u][r] + bv;
                    if (EPI == 1) v = 0.5f * v * (1.0f + erff(v * 0.70710678118654752f));
                    const long long cidx = cbase + (long long)cm * N + cn;
                    if (EPI == 2) v += residual[cidx];
                    if (C)   C[cidx] = v;
                    if (Cbf) Cbf[cidx] = (__bf16)v;
                }
            }
        }
    }
}

// ---------------------------------------------------------------------------
// scores = (p==r) ? -1e9 : scores + pos_bias; then *= 1/TEMP
// ---------------------------------------------------------------------------
__global__ void scores_post_kernel(float* __restrict__ scores,
                                   const float* __restrict__ pos) {
    const long long i = (long long)blockIdx.x * 256 + threadIdx.x;
    const long long PP = (long long)P_ * P_;
    if (i >= (long long)B_ * PP) return;
    const int pr = (int)(i % PP);
    const int prow = pr / P_, pcol = pr % P_;
    float v = (prow == pcol) ? -1e9f : (scores[i] + pos[pr]);
    scores[i] = v * INV_TEMP_;
}

// ---------------------------------------------------------------------------
// Top-32 of 576 per row (one wave32 per row) + softmax -> routes, log-weights
// ---------------------------------------------------------------------------
__global__ __launch_bounds__(256) void topk_kernel(
    const float* __restrict__ scores, int* __restrict__ routes,
    float* __restrict__ logw) {
    __shared__ float srow[8][P_];
    __shared__ float tv[8][KW_];
    const int wave = threadIdx.x >> 5, lane = threadIdx.x & 31;
    const long long row = (long long)blockIdx.x * 8 + wave;   // 0..B*P-1
    const float* sr = scores + row * P_;
    for (int i = lane; i < P_; i += 32) srow[wave][i] = sr[i];
    __syncthreads();

    for (int t = 0; t < KW_; ++t) {
        float bv = -3.4e38f; int bi = 0;
        for (int i = lane; i < P_; i += 32) {
            const float v = srow[wave][i];
            if (v > bv) { bv = v; bi = i; }
        }
#pragma unroll
        for (int o = 16; o > 0; o >>= 1) {
            const float ov = __shfl_xor(bv, o);
            const int   oi = __shfl_xor(bi, o);
            if (ov > bv || (ov == bv && oi < bi)) { bv = ov; bi = oi; }
        }
        if (lane == 0) {
            routes[row * KW_ + t] = bi;
            tv[wave][t] = bv;
            srow[wave][bi] = -3.4e38f;
        }
        __syncthreads();
    }
    // softmax over the 32 selected scores (descending order, matches top_k)
    const float v = tv[wave][lane];
    float mx = v;
#pragma unroll
    for (int o = 16; o > 0; o >>= 1) mx = fmaxf(mx, __shfl_xor(mx, o));
    const float e = __expf(v - mx);
    float sum = e;
#pragma unroll
    for (int o = 16; o > 0; o >>= 1) sum += __shfl_xor(sum, o);
    const float w = e / sum;
    logw[row * KW_ + lane] = fmaxf(__logf(w), -10.0f);
}

// ---------------------------------------------------------------------------
// CLS token full attention: one block per (b,h); 577 keys, HD=64
// ---------------------------------------------------------------------------
__global__ __launch_bounds__(256) void cls_attn_kernel(
    const float* __restrict__ qkv, float* __restrict__ out) {
    const int bh = blockIdx.x;                // 0..B*H-1
    const int b = bh / H_, h = bh % H_;
    const int tid = threadIdx.x;
    __shared__ float qv[HD_];
    __shared__ float prob[S_];
    __shared__ float red[8];
    __shared__ float part[4][HD_];
    const long long base = (long long)b * S_ * D3_;
    if (tid < HD_) qv[tid] = qkv[base + h * HD_ + tid];
    __syncthreads();

    float lmax = -3.4e38f;
    for (int j = tid; j < S_; j += 256) {
        const float* kp = qkv + base + (long long)j * D3_ + D_ + h * HD_;
        float d = 0.f;
#pragma unroll 16
        for (int c = 0; c < HD_; ++c) d += qv[c] * kp[c];
        d *= SCALE_;
        prob[j] = d;
        lmax = fmaxf(lmax, d);
    }
#pragma unroll
    for (int o = 16; o > 0; o >>= 1) lmax = fmaxf(lmax, __shfl_xor(lmax, o));
    const int wave = tid >> 5, lane = tid & 31;
    if (lane == 0) red[wave] = lmax;
    __syncthreads();
    float m = red[0];
#pragma unroll
    for (int i = 1; i < 8; ++i) m = fmaxf(m, red[i]);
    __syncthreads();

    float ls = 0.f;
    for (int j = tid; j < S_; j += 256) {
        const float e = __expf(prob[j] - m);
        prob[j] = e;
        ls += e;
    }
#pragma unroll
    for (int o = 16; o > 0; o >>= 1) ls += __shfl_xor(ls, o);
    if (lane == 0) red[wave] = ls;
    __syncthreads();
    float sum = 0.f;
#pragma unroll
    for (int i = 0; i < 8; ++i) sum += red[i];
    const float inv = 1.0f / sum;
    __syncthreads();

    const int gidx = tid >> 6, d = tid & 63;   // 4 key-groups x 64 dims
    float accv = 0.f;
    for (int j = gidx; j < S_; j += 4)
        accv += prob[j] * qkv[base + (long long)j * D3_ + 2 * D_ + h * HD_ + d];
    part[gidx][d] = accv;
    __syncthreads();
    if (gidx == 0) {
        const float o4 = part[0][d] + part[1][d] + part[2][d] + part[3][d];
        out[((long long)b * S_) * D_ + h * HD_ + d] = o4 * inv;
    }
}

// ---------------------------------------------------------------------------
// Gathered patch attention: one wave32 per (b,h,p); lane = routed key slot
// ---------------------------------------------------------------------------
__global__ __launch_bounds__(256) void patch_attn_kernel(
    const float* __restrict__ qkv, const int* __restrict__ routes,
    const float* __restrict__ logw, float* __restrict__ out) {
    const int wave = threadIdx.x >> 5, lane = threadIdx.x & 31;
    const long long rid = (long long)blockIdx.x * 8 + wave;  // 0..B*H*P-1
    const int b = (int)(rid / (H_ * P_));
    const int hp = (int)(rid % (H_ * P_));
    const int h = hp / P_, pp = hp % P_;
    __shared__ float qs[8][HD_];
    __shared__ float aw[8][KW_];
    __shared__ int   rt[8][KW_];

    const long long tokbase = ((long long)b * S_ + pp + 1) * D3_;
    qs[wave][lane]      = qkv[tokbase + h * HD_ + lane];
    qs[wave][lane + 32] = qkv[tokbase + h * HD_ + lane + 32];
    const long long ridx = ((long long)b * P_ + pp) * KW_ + lane;
    const int r = routes[ridx];
    rt[wave][lane] = r;
    __syncthreads();

    const float* kp = qkv + ((long long)b * S_ + r + 1) * D3_ + D_ + h * HD_;
    float d = 0.f;
#pragma unroll 16
    for (int c = 0; c < HD_; ++c) d += qs[wave][c] * kp[c];
    float logit = d * SCALE_ + logw[ridx];
    float mx = logit;
#pragma unroll
    for (int o = 16; o > 0; o >>= 1) mx = fmaxf(mx, __shfl_xor(mx, o));
    const float e = __expf(logit - mx);
    float s = e;
#pragma unroll
    for (int o = 16; o > 0; o >>= 1) s += __shfl_xor(s, o);
    aw[wave][lane] = e / s;
    __syncthreads();

    float o0 = 0.f, o1 = 0.f;
#pragma unroll 4
    for (int kk = 0; kk < KW_; ++kk) {
        const float a = aw[wave][kk];
        const float* vp = qkv + ((long long)b * S_ + rt[wave][kk] + 1) * D3_ +
                          2 * D_ + h * HD_;
        o0 += a * vp[lane];
        o1 += a * vp[lane + 32];
    }
    const long long ob = ((long long)b * S_ + pp + 1) * D_ + h * HD_;
    out[ob + lane]      = o0;
    out[ob + lane + 32] = o1;
}

// ---------------------------------------------------------------------------
// Host-side orchestration
// ---------------------------------------------------------------------------
extern "C" void kernel_launch(void* const* d_in, const int* in_sizes, int n_in,
                              void* d_out, int out_size, void* d_ws, size_t ws_size,
                              hipStream_t stream) {
    const float* x      = (const float*)d_in[0];
    const float* g1     = (const float*)d_in[1];
    const float* b1     = (const float*)d_in[2];
    const float* wq     = (const float*)d_in[3];
    const float* bq     = (const float*)d_in[4];
    const float* wk     = (const float*)d_in[5];
    const float* bk     = (const float*)d_in[6];
    const float* posb   = (const float*)d_in[7];
    const float* w_qkv  = (const float*)d_in[8];
    const float* b_qkv  = (const float*)d_in[9];
    const float* w_proj = (const float*)d_in[10];
    const float* b_proj = (const float*)d_in[11];
    const float* g2     = (const float*)d_in[12];
    const float* b2     = (const float*)d_in[13];
    const float* w_mlp1 = (const float*)d_in[14];
    const float* b_mlp1 = (const float*)d_in[15];
    const float* w_mlp2 = (const float*)d_in[16];
    const float* b_mlp2 = (const float*)d_in[17];
    float* out = (float*)d_out;

    // ---- workspace carve-up (deterministic) ----
    char* p = (char*)d_ws;
    auto alloc = [&](size_t bytes) -> void* {
        void* r = (void*)p;
        p += (bytes + 255) & ~(size_t)255;
        return r;
    };
    float*  xn       = (float*) alloc(sizeof(float) * NS_ * D_);
    __bf16* xn_bf    = (__bf16*)alloc(2ull * NS_ * D_);
    __bf16* wqT      = (__bf16*)alloc(2ull * D_ * D_);       // N x K
    __bf16* wkT      = (__bf16*)alloc(2ull * D_ * D_);
    __bf16* wqkvT    = (__bf16*)alloc(2ull * D3_ * D_);
    __bf16* wprojT   = (__bf16*)alloc(2ull * D_ * D_);
    __bf16* wm1T     = (__bf16*)alloc(2ull * DM_ * D_);
    __bf16* wm2T     = (__bf16*)alloc(2ull * D_ * DM_);
    float*  tmp_full = (float*) alloc(sizeof(float) * NS_ * D_);
    __bf16* q_bf     = (__bf16*)alloc(2ull * B_ * P_ * D_);
    __bf16* k_bf     = (__bf16*)alloc(2ull * B_ * P_ * D_);
    float*  scores   = (float*) alloc(sizeof(float) * B_ * P_ * P_);
    int*    routes   = (int*)   alloc(sizeof(int)   * B_ * P_ * KW_);
    float*  logw     = (float*) alloc(sizeof(float) * B_ * P_ * KW_);
    float*  qkv      = (float*) alloc(sizeof(float) * NS_ * D3_);
    float*  attn     = (float*) alloc(sizeof(float) * NS_ * D_);
    __bf16* attn_bf  = (__bf16*)alloc(2ull * NS_ * D_);
    float*  x1       = (float*) alloc(sizeof(float) * NS_ * D_);
    __bf16* hln_bf   = (__bf16*)alloc(2ull * NS_ * D_);
    __bf16* hmid_bf  = (__bf16*)alloc(2ull * NS_ * DM_);
    (void)ws_size; (void)in_sizes; (void)n_in; (void)out_size;

    auto conv = [&](const float* src, __bf16* dst, long long n) {
        f32_to_bf16_kernel<<<dim3((unsigned)((n + 255) / 256)), 256, 0, stream>>>(src, dst, n);
    };
    auto transp = [&](const float* src, __bf16* dst, int K, int N) {
        transpose_bf16_kernel<<<dim3(N / 32, K / 32), dim3(32, 8), 0, stream>>>(src, dst, K, N);
    };

    // 1) weights -> bf16, transposed to (N x K) so every GEMM is A @ Bt^T
    transp(wq,     wqT,    D_,  D_);
    transp(wk,     wkT,    D_,  D_);
    transp(w_qkv,  wqkvT,  D_,  D3_);
    transp(w_proj, wprojT, D_,  D_);
    transp(w_mlp1, wm1T,   D_,  DM_);
    transp(w_mlp2, wm2T,   DM_, D_);

    const int MB64 = (NS_ + 63) / 64;   // 73 row-tiles (BM=64 config)

    // 2) LN1
    layernorm_kernel<<<NS_, 256, 0, stream>>>(x, g1, b1, xn, xn_bf);

    // 3) q = l2norm(xn @ wq + bq)   (compute on all rows; pack drops s==0)
    gemm_bt_wmma<2, 2, 0><<<dim3(D_ / 128, MB64, 1), 256, 0, stream>>>(
        xn_bf, wqT, bq, nullptr, tmp_full, nullptr, NS_, D_, D_, 0, 0, 0);
    l2norm_pack_kernel<<<B_ * P_, 256, 0, stream>>>(tmp_full, q_bf);

    // 4) k = l2norm(xn @ wk + bk)
    gemm_bt_wmma<2, 2, 0><<<dim3(D_ / 128, MB64, 1), 256, 0, stream>>>(
        xn_bf, wkT, bk, nullptr, tmp_full, nullptr, NS_, D_, D_, 0, 0, 0);
    l2norm_pack_kernel<<<B_ * P_, 256, 0, stream>>>(tmp_full, k_bf);

    // 5) routing scores: per-batch q @ k^T  (batched GEMM, z = B; 32x64 tiles)
    gemm_bt_wmma<1, 1, 0><<<dim3(P_ / 64, P_ / 32, B_), 256, 0, stream>>>(
        q_bf, k_bf, nullptr, nullptr, scores, nullptr, P_, P_, D_,
        (long long)P_ * D_, (long long)P_ * D_, (long long)P_ * P_);
    {
        const long long n = (long long)B_ * P_ * P_;
        scores_post_kernel<<<dim3((unsigned)((n + 255) / 256)), 256, 0, stream>>>(scores, posb);
    }

    // 6) top-32 routing + softmax weights
    topk_kernel<<<(B_ * P_) / 8, 256, 0, stream>>>(scores, routes, logw);

    // 7) fused QKV projection
    gemm_bt_wmma<2, 2, 0><<<dim3(D3_ / 128, MB64, 1), 256, 0, stream>>>(
        xn_bf, wqkvT, b_qkv, nullptr, qkv, nullptr, NS_, D3_, D_, 0, 0, 0);

    // 8) attention (CLS dense + routed sparse)
    cls_attn_kernel<<<B_ * H_, 256, 0, stream>>>(qkv, attn);
    patch_attn_kernel<<<(B_ * H_ * P_) / 8, 256, 0, stream>>>(qkv, routes, logw, attn);
    conv(attn, attn_bf, (long long)NS_ * D_);

    // 9) output projection + residual -> x1
    gemm_bt_wmma<2, 2, 2><<<dim3(D_ / 128, MB64, 1), 256, 0, stream>>>(
        attn_bf, wprojT, b_proj, x, x1, nullptr, NS_, D_, D_, 0, 0, 0);

    // 10) LN2 -> bf16 only
    layernorm_kernel<<<NS_, 256, 0, stream>>>(x1, g2, b2, nullptr, hln_bf);

    // 11) MLP1 with exact GELU, bf16 out only
    gemm_bt_wmma<2, 2, 1><<<dim3(DM_ / 128, MB64, 1), 256, 0, stream>>>(
        hln_bf, wm1T, b_mlp1, nullptr, nullptr, hmid_bf, NS_, DM_, D_, 0, 0, 0);

    // 12) MLP2 + residual -> final output
    gemm_bt_wmma<2, 2, 2><<<dim3(D_ / 128, MB64, 1), 256, 0, stream>>>(
        hmid_bf, wm2T, b_mlp2, x1, out, nullptr, NS_, D_, DM_, 0, 0, 0);
}
